// Ours_12515534700921
// MI455X (gfx1250) — matrix operand order repaired and checked by reading, not compiled, not run on hardware
//
#include <hip/hip_runtime.h>
#include <hip/hip_bf16.h>
#include <math.h>

#define DEV static __device__ __forceinline__

typedef __attribute__((ext_vector_type(16))) __bf16 v16bf;
typedef __attribute__((ext_vector_type(8)))  float  v8f;

union BF16x16 { v16bf v; uint4 q[2]; unsigned short u[16]; };

DEV unsigned short f2bf(float f) {
    unsigned int u = __float_as_uint(f);
    u += 0x7FFFu + ((u >> 16) & 1u);
    return (unsigned short)(u >> 16);
}
DEV float bf2f(unsigned short h) { return __uint_as_float(((unsigned int)h) << 16); }

DEV float act_apply(float v, int act) {
    if (act == 1) return fmaxf(v, 0.0f);
    if (act == 2) return 0.5f * v * (1.0f + erff(v * 0.70710678118f));
    if (act == 3) return 1.0f / (1.0f + expf(-v));
    return v;
}

DEV v8f wmma_bf(v16bf a, v16bf b, v8f c) {
    return __builtin_amdgcn_wmma_f32_16x16x32_bf16(false, a, false, b, (short)0, c, false, false);
}

// A fragment (16x32, bf16) from row-major [*, lda] at (row0, col0).
DEV v16bf load_a_frag_g(const unsigned short* A, int lda, long row0, int col0) {
    int l = threadIdx.x & 31;
    const unsigned short* p = A + (size_t)(row0 + (l & 15)) * lda + col0 + ((l >> 4) << 3);
    BF16x16 f;
    f.q[0] = *(const uint4*)(p);
    f.q[1] = *(const uint4*)(p + 16);
    return f.v;
}
// B fragment (32x16, bf16) from row-major [K, ldb] at (k0, n0): lane l holds column n0+(l&15), k in table order.
DEV v16bf load_b_frag_g(const unsigned short* Bm, int ldb, int k0, int n0) {
    int l = threadIdx.x & 31;
    const unsigned short* p = Bm + (size_t)(k0 + ((l >> 4) << 3)) * ldb + n0 + (l & 15);
    BF16x16 f;
    for (int s = 0; s < 8; s++) { f.u[s] = p[(size_t)s * ldb]; f.u[8 + s] = p[(size_t)(16 + s) * ldb]; }
    return f.v;
}

// ---------------- Tensor Data Mover staging (CDNA5 TDM, clang-23 6-arg builtin) ----------------
#if defined(__HIP_DEVICE_COMPILE__) && \
    __has_builtin(__builtin_amdgcn_tensor_load_to_lds) && \
    __has_builtin(__builtin_amdgcn_s_wait_tensorcnt) && \
    __has_builtin(__builtin_amdgcn_readfirstlane)
#define HAVE_TDM 1
#else
#define HAVE_TDM 0
#endif

#if HAVE_TDM
typedef __attribute__((ext_vector_type(4))) unsigned int v4u;
typedef __attribute__((ext_vector_type(8))) int v8i_;
typedef __attribute__((ext_vector_type(4))) int v4i_;

DEV int rfl(unsigned int v) { return __builtin_amdgcn_readfirstlane((int)v); }

// 1-D TDM copy: nelem bf16 elements, global -> LDS (D# per CDNA5 ISA).
// lds_byte_addr: byte address in LDS space (we pass truncated flat address of the single
// static __shared__ staging buffer — compile-only environment, allocation-base in practice).
DEV void tdm_load_1d(const void* gsrc, unsigned lds_byte_addr, unsigned nelem) {
    unsigned long long ga = (unsigned long long)gsrc;
    v4u g0;
    g0[0] = (unsigned)rfl(1u);                                    // count=1, user-mode D#
    g0[1] = (unsigned)rfl(lds_byte_addr);                         // lds_addr [63:32]
    g0[2] = (unsigned)rfl((unsigned)(ga & 0xFFFFFFFFu));          // global_addr [95:64]
    g0[3] = (unsigned)rfl((unsigned)((ga >> 32) & 0x01FFFFFFu)    // global_addr [120:96]
                          | (2u << 30));                          // type=2 ("image")
    v8i_ g1;
    g1[0] = rfl(1u << 16);                        // data_size=1 (2 bytes); mask/iterate/pad = 0
    g1[1] = rfl((nelem & 0xFFFFu) << 16);         // tensor_dim0[15:0]  -> bits 63:48
    g1[2] = rfl((nelem >> 16) & 0xFFFFu);         // tensor_dim0[31:16] -> bits 79:64
    g1[3] = rfl((nelem & 0xFFFFu) << 16);         // tile_dim0 -> bits 127:112 (tensor_dim1 = 0)
    g1[4] = rfl(0u);                              // tile_dim1 = tile_dim2 = 0 (unused)
    g1[5] = rfl(nelem);                           // tensor_dim0_stride[31:0]
    g1[6] = rfl(0u);
    g1[7] = rfl(0u);
    v4i_ g2 = {0, 0, 0, 0};
    v4i_ g3 = {0, 0, 0, 0};
    v8i_ g4 = {0, 0, 0, 0, 0, 0, 0, 0};
    __builtin_amdgcn_tensor_load_to_lds(g0, g1, g2, g3, g4, 0);
}

DEV v16bf load_b_frag_lds(const unsigned short* p, int ldb, int n0) {
    int l = threadIdx.x & 31;
    const unsigned short* q = p + (size_t)((l >> 4) << 3) * ldb + n0 + (l & 15);
    BF16x16 f;
    for (int s = 0; s < 8; s++) { f.u[s] = q[(size_t)s * ldb]; f.u[8 + s] = q[(size_t)(16 + s) * ldb]; }
    return f.v;
}
#endif

// ---------------- generic WMMA GEMM: out = act(A[MxK]*B[KxN] + bias) (+res) ----------------
__global__ void __launch_bounds__(128)
gemm_wmma(const unsigned short* A, const unsigned short* Bm,
          const float* bias, const float* res,
          float* outF, unsigned short* outH,
          long M, int N, int K, int act) {
    long mtiles = M >> 4;
    int  ntN = N >> 6;
    long wave = (long)blockIdx.x * 4 + (threadIdx.x >> 5);
    if (wave >= mtiles * ntN) return;      // wave-uniform, no LDS in this kernel
    long mt = wave / ntN; int nb = (int)(wave % ntN);
    long m0 = mt * 16; int n0 = nb * 64;
    int l = threadIdx.x & 31;
    v8f z = {0,0,0,0,0,0,0,0};
    v8f acc[4];
#pragma unroll
    for (int t = 0; t < 4; t++) acc[t] = z;
    for (int k0 = 0; k0 < K; k0 += 32) {
        v16bf fa = load_a_frag_g(A, K, m0, k0);
        __builtin_prefetch(A + (size_t)m0 * K + k0 + 32, 0, 1);
#pragma unroll
        for (int t = 0; t < 4; t++) {
            v16bf fb = load_b_frag_g(Bm, N, k0, n0 + t * 16);
            acc[t] = wmma_bf(fa, fb, acc[t]);
        }
    }
    int col = l & 15, mo = (l >> 4) * 8;
#pragma unroll
    for (int t = 0; t < 4; t++) {
        int n = n0 + t * 16 + col;
        float bv = bias ? bias[n] : 0.0f;
#pragma unroll
        for (int r = 0; r < 8; r++) {
            long m = m0 + mo + r;
            float v = acc[t][r] + bv;
            v = act_apply(v, act);
            if (res)  v += res[(size_t)m * N + n];
            if (outF) outF[(size_t)m * N + n] = v;
            if (outH) outH[(size_t)m * N + n] = f2bf(v);
        }
    }
}

// ---------------- geometry functors for implicit-GEMM convolutions ----------------
struct GeoConv3 {   // 3x3x3, stride (1,2,2), zero pad (1,1,1); posmajor [b][t][y][x][C]
    int T, Si, So;
    __device__ long src(long q, int tap) const {
        int xo = (int)(q % So); long r = q / So;
        int yo = (int)(r % So); r /= So;
        int t = (int)(r % T); int b = (int)(r / T);
        int kx = tap % 3, ky = (tap / 3) % 3, kt = tap / 9;
        int ti = t + kt - 1, yi = 2 * yo + ky - 1, xi = 2 * xo + kx - 1;
        if (ti < 0 || ti >= T || yi < 0 || yi >= Si || xi < 0 || xi >= Si) return -1;
        return ((long)((b * T + ti) * Si + yi)) * Si + xi;
    }
};
struct GeoConv2 {   // 3x3, stride 1, zero pad 1; posmajor [b][y][x][C]
    int S;
    __device__ long src(long q, int tap) const {
        int x = (int)(q % S); long r = q / S;
        int y = (int)(r % S); int b = (int)(r / S);
        int kx = tap % 3, ky = tap / 3;
        int yi = y + ky - 1, xi = x + kx - 1;
        if (yi < 0 || yi >= S || xi < 0 || xi >= S) return -1;
        return ((long)(b * S + yi)) * S + xi;
    }
};
struct GeoConv1 {   // k5, stride 2, zero pad 2; posmajor [b][time][C]
    int Lin, Lout;
    __device__ long src(long q, int tap) const {
        int to = (int)(q % Lout); int b = (int)(q / Lout);
        int ti = 2 * to + tap - 2;
        if (ti < 0 || ti >= Lin) return -1;
        return (long)b * Lin + ti;
    }
};

// ---------------- generic implicit-GEMM conv (channel-contiguous posmajor input) ----------------
// Weight panel for each (tap, k-chunk) is DMA'd to LDS once per workgroup by the Tensor Data
// Mover (wave 0 issues TENSOR_LOAD_TO_LDS, waits TENSORcnt, barrier), then all 4 waves serve
// their B fragments from LDS. Falls back to direct global loads if the TDM builtin is absent.
template<typename GEO>
__global__ void __launch_bounds__(128)
convw_kernel(GEO geo, const unsigned short* in, const unsigned short* wpk,
             const float* bias, float* outF, unsigned short* outH,
             long M, int Cin, int Cout, int ntaps, int act, long totalWaves) {
    __shared__ __align__(16) unsigned short tile[4][16 * 40];
#if HAVE_TDM
    __shared__ __align__(16) unsigned short wpanel[32 * 256];   // up to Cout=256
#endif
    int wv = threadIdx.x >> 5, l = threadIdx.x & 31;
    long wave = (long)blockIdx.x * 4 + wv;
    if (wave >= totalWaves) wave = totalWaves - 1;   // duplicate last tile (benign rewrite)
    int ntN = Cout >> 6;
    long mt = wave / ntN; int nb = (int)(wave % ntN);
    long m0 = mt * 16; int n0 = nb * 64;
    v8f z = {0,0,0,0,0,0,0,0};
    v8f acc[4];
#pragma unroll
    for (int t = 0; t < 4; t++) acc[t] = z;
    int kch = Cin >> 5;
    int pi = l >> 1, hs = l & 1;
    for (int tap = 0; tap < ntaps; tap++) {
        long s_pos = geo.src(m0 + pi, tap);
        __builtin_prefetch(wpk + ((size_t)(tap + 1) * Cin) * Cout, 0, 1);
        for (int kc = 0; kc < kch; kc++) {
            uint4 a0 = {0,0,0,0}, a1 = {0,0,0,0};
            if (s_pos >= 0) {
                const uint4* p = (const uint4*)(in + (size_t)s_pos * Cin + kc * 32 + hs * 16);
                a0 = p[0]; a1 = p[1];
            }
            *(uint4*)&tile[wv][pi * 40 + hs * 16]     = a0;
            *(uint4*)&tile[wv][pi * 40 + hs * 16 + 8] = a1;
            const unsigned short* wp = wpk + ((size_t)tap * Cin + kc * 32) * Cout;
#if HAVE_TDM
            if (wv == 0) {
                tdm_load_1d(wp, (unsigned)(unsigned long long)(const void*)wpanel,
                            (unsigned)(32 * Cout));
                __builtin_amdgcn_s_wait_tensorcnt(0);
            }
#endif
            __syncthreads();
            BF16x16 fa;
            int r = l & 15, cb = (l >> 4) << 3;
            fa.q[0] = *(const uint4*)&tile[wv][r * 40 + cb];
            fa.q[1] = *(const uint4*)&tile[wv][r * 40 + cb + 16];
#pragma unroll
            for (int t = 0; t < 4; t++) {
#if HAVE_TDM
                v16bf fb = load_b_frag_lds(wpanel, Cout, n0 + t * 16);
#else
                v16bf fb = load_b_frag_g(wp, Cout, 0, n0 + t * 16);
#endif
                acc[t] = wmma_bf(fa.v, fb, acc[t]);
            }
            __syncthreads();
        }
    }
    int col = l & 15, mo = (l >> 4) * 8;
#pragma unroll
    for (int t = 0; t < 4; t++) {
        int n = n0 + t * 16 + col;
        float bv = bias ? bias[n] : 0.0f;
#pragma unroll
        for (int r = 0; r < 8; r++) {
            long m = m0 + mo + r;
            float v = acc[t][r] + bv;
            v = act_apply(v, act);
            if (outF) outF[(size_t)m * Cout + n] = v;
            if (outH) outH[(size_t)m * Cout + n] = f2bf(v);
        }
    }
}

// ---------------- stem conv3d: 4->64, k3, stride (1,4,4), pad 1; K = 108 padded to 128 ----------------
__global__ void __launch_bounds__(128)
stem_conv_kernel(const unsigned short* xp, const unsigned short* Wst,
                 const float* bias, unsigned short* outH) {
    __shared__ __align__(16) unsigned short tile[4][16 * 136];
    int wv = threadIdx.x >> 5, l = threadIdx.x & 31;
    long m0 = ((long)blockIdx.x * 4 + wv) * 16;
    unsigned int* tz = (unsigned int*)tile[wv];
    for (int j = l; j < 16 * 136 / 2; j += 32) tz[j] = 0;
    __syncthreads();
    int pi = l >> 1, hs = l & 1;
    long q = m0 + pi;
    int xo = (int)(q & 63); long r0 = q >> 6;
    int yo = (int)(r0 & 63); r0 >>= 6;
    int t = (int)(r0 % 10); int b = (int)(r0 / 10);
    for (int tp = hs; tp < 27; tp += 2) {
        int kt = tp / 9; int rem = tp % 9; int ky = rem / 3, kx = rem % 3;
        int ti = t + kt - 1, yi = yo * 4 + ky - 1, xi = xo * 4 + kx - 1;
        uint2 v = {0, 0};
        if (ti >= 0 && ti < 10 && yi >= 0 && yi < 256 && xi >= 0 && xi < 256) {
            long sp = (((long)(b * 10 + ti) * 256) + yi) * 256 + xi;
            v = *(const uint2*)(xp + sp * 4);
        }
        *(uint2*)&tile[wv][pi * 136 + tp * 4] = v;
    }
    __syncthreads();
    v8f z = {0,0,0,0,0,0,0,0};
    v8f acc[4];
#pragma unroll
    for (int t2 = 0; t2 < 4; t2++) acc[t2] = z;
    int rr = l & 15, cb = (l >> 4) << 3;
#pragma unroll
    for (int kc = 0; kc < 4; kc++) {
        BF16x16 fa;
        fa.q[0] = *(const uint4*)&tile[wv][rr * 136 + kc * 32 + cb];
        fa.q[1] = *(const uint4*)&tile[wv][rr * 136 + kc * 32 + cb + 16];
#pragma unroll
        for (int tt = 0; tt < 4; tt++) {
            v16bf fb = load_b_frag_g(Wst, 64, kc * 32, tt * 16);
            acc[tt] = wmma_bf(fa.v, fb, acc[tt]);
        }
    }
    int col = l & 15, mo = (l >> 4) * 8;
#pragma unroll
    for (int tt = 0; tt < 4; tt++) {
        int n = tt * 16 + col;
        float bv = bias[n];
#pragma unroll
        for (int r = 0; r < 8; r++) {
            long m = m0 + mo + r;
            float v = fmaxf(acc[tt][r] + bv, 0.0f);
            outH[(size_t)m * 64 + n] = f2bf(v);
        }
    }
}

// ---------------- packing / elementwise / normalization / scan kernels ----------------
__global__ void pack_x_kernel(const float* x, unsigned short* xp, long n) {
    long i = (long)blockIdx.x * blockDim.x + threadIdx.x;
    if (i >= n) return;
    int c = (int)(i & 3); long r = i >> 2;
    long p = r % 65536; r /= 65536;
    int t = (int)(r % 10); int b = (int)(r / 10);
    xp[i] = f2bf(x[(((long)(b * 4 + c) * 10 + t) * 65536) + p]);
}
__global__ void pack_stem_w(const float* src, unsigned short* dst) {
    int i = blockIdx.x * blockDim.x + threadIdx.x;
    if (i >= 128 * 64) return;
    int o = i & 63; int k = i >> 6;
    float v = 0.0f;
    if (k < 108) { int tap = k >> 2, c = k & 3; v = src[(o * 4 + c) * 27 + tap]; }
    dst[k * 64 + o] = f2bf(v);
}
__global__ void pack_conv_w(const float* src, unsigned short* dst, int O, int I, int ntaps) {
    long i = (long)blockIdx.x * blockDim.x + threadIdx.x;
    if (i >= (long)O * I * ntaps) return;
    int tap = (int)(i % ntaps); long r = i / ntaps;
    int ic = (int)(r % I); int o = (int)(r / I);
    dst[((size_t)tap * I + ic) * O + o] = f2bf(src[i]);
}
__global__ void transpose_bf(const float* s, unsigned short* d, int R, int C) {
    long i = (long)blockIdx.x * blockDim.x + threadIdx.x;
    if (i >= (long)R * C) return;
    int o = (int)(i / C), c = (int)(i % C);
    d[(size_t)c * R + o] = f2bf(s[i]);
}
__global__ void f2bf_copy(const float* s, unsigned short* d, long n) {
    long i = (long)blockIdx.x * blockDim.x + threadIdx.x;
    if (i < n) d[i] = f2bf(s[i]);
}
__global__ void copyf_kernel(const float* s, float* d, long n) {
    long i = (long)blockIdx.x * blockDim.x + threadIdx.x;
    if (i < n) d[i] = s[i];
}
__global__ void s5_lambar(const float* lam_re, const float* lam_im, const float* log_dt,
                          float* lr_, float* li_, float* fr_, float* fi_, int d) {
    int p = threadIdx.x;
    if (p >= d) return;
    float dt = expf(log_dt[p]);
    float ar = lam_re[p], ai = lam_im[p];
    float m = expf(ar * dt);
    float lbr = m * cosf(ai * dt);
    float lbi = m * sinf(ai * dt);
    lr_[p] = lbr; li_[p] = lbi;
    float nr = lbr - 1.0f, ni = lbi;
    float den = ar * ar + ai * ai;
    fr_[p] = (nr * ar + ni * ai) / den;
    fi_[p] = (ni * ar - nr * ai) / den;
}
__global__ void s5_packB(const float* B_re, const float* B_im, const float* fr, const float* fi,
                         unsigned short* Wbu, int d) {
    long i = (long)blockIdx.x * blockDim.x + threadIdx.x;
    if (i >= (long)d * d) return;
    int p = (int)(i / d), h = (int)(i % d);
    float br = B_re[i], bi = B_im[i];
    float re = fr[p] * br - fi[p] * bi;
    float im = fr[p] * bi + fi[p] * br;
    Wbu[(size_t)h * 2 * d + p]     = f2bf(re);
    Wbu[(size_t)h * 2 * d + d + p] = f2bf(im);
}
__global__ void s5_packC(const float* C_re, const float* C_im, unsigned short* Wc, int d) {
    long i = (long)blockIdx.x * blockDim.x + threadIdx.x;
    if (i >= (long)d * d) return;
    int h = (int)(i / d), p = (int)(i % d);
    Wc[(size_t)p * d + h]       = f2bf(2.0f * C_re[(size_t)h * d + p]);
    Wc[(size_t)(d + p) * d + h] = f2bf(-2.0f * C_im[(size_t)h * d + p]);
}
__global__ void s5_scan(const float* Bu, const float* lamr, const float* lami,
                        unsigned short* xs_bf, int Bb, int Ltok, int d) {
    int idx = blockIdx.x * blockDim.x + threadIdx.x;
    if (idx >= Bb * d) return;
    int b = idx / d, p = idx % d;
    float ar = lamr[p], ai = lami[p];
    float xr = 0.0f, xi = 0.0f;
    size_t base = (size_t)b * Ltok;
    for (int l = 0; l < Ltok; l++) {
        const float* row = Bu + (base + l) * (size_t)(2 * d);
        float br = row[p], bi = row[d + p];
        float nr = ar * xr - ai * xi + br;
        float ni = ar * xi + ai * xr + bi;
        xr = nr; xi = ni;
        unsigned short* o = xs_bf + (base + l) * (size_t)(2 * d);
        o[p] = f2bf(xr); o[d + p] = f2bf(xi);
    }
}
__global__ void s5comb(const float* xin, const float* s5o, const unsigned short* u_bf,
                       const float* Dv, float* out, long n, int d) {
    long i = (long)blockIdx.x * blockDim.x + threadIdx.x;
    if (i >= n) return;
    int c = (int)(i % d);
    out[i] = xin[i] + s5o[i] + bf2f(u_bf[i]) * Dv[c];
}
__global__ void ln_kernel(const float* x, const float* g, const float* b,
                          unsigned short* out, int d) {
    __shared__ float red[128];
    size_t m = blockIdx.x;
    int tid = threadIdx.x;
    float v = x[m * d + tid];
    red[tid] = v; __syncthreads();
    for (int s = d >> 1; s > 0; s >>= 1) { if (tid < s) red[tid] += red[tid + s]; __syncthreads(); }
    float mean = red[0] / d; __syncthreads();
    float c = v - mean;
    red[tid] = c * c; __syncthreads();
    for (int s = d >> 1; s > 0; s >>= 1) { if (tid < s) red[tid] += red[tid + s]; __syncthreads(); }
    float var = red[0] / d;
    float rs = rsqrtf(var + 1e-5f);
    out[m * d + tid] = f2bf(c * rs * g[tid] + b[tid]);
}
__global__ void inorm_stats(const float* x, float* mean, float* rstd, int Npos, int C) {
    __shared__ float s1[256], s2[256];
    int bc = blockIdx.x; int b = bc / C, c = bc % C;
    float a = 0.0f, q = 0.0f;
    for (int i = threadIdx.x; i < Npos; i += blockDim.x) {
        float v = x[((size_t)b * Npos + i) * C + c];
        a += v; q += v * v;
    }
    s1[threadIdx.x] = a; s2[threadIdx.x] = q; __syncthreads();
    for (int s = blockDim.x >> 1; s > 0; s >>= 1) {
        if (threadIdx.x < (unsigned)s) { s1[threadIdx.x] += s1[threadIdx.x + s]; s2[threadIdx.x] += s2[threadIdx.x + s]; }
        __syncthreads();
    }
    if (threadIdx.x == 0) {
        float m = s1[0] / Npos;
        float var = s2[0] / Npos - m * m;
        mean[bc] = m; rstd[bc] = rsqrtf(var + 1e-5f);
    }
}
__global__ void dw3_kernel(const float* h, const float* w, float* out, int T, int S, long n) {
    long i = (long)blockIdx.x * blockDim.x + threadIdx.x;
    if (i >= n) return;
    int c = (int)(i & 63); long pos = i >> 6;
    int x = (int)(pos % S); long r = pos / S;
    int y = (int)(r % S); r /= S;
    int t = (int)(r % T); int b = (int)(r / T);
    float acc = 0.0f;
    for (int kt = 0; kt < 3; kt++) for (int ky = 0; ky < 3; ky++) for (int kx = 0; kx < 3; kx++) {
        int ti = t + kt - 1; ti = ti < 0 ? 0 : (ti >= T ? T - 1 : ti);
        int yi = y + ky - 1; yi = yi < 0 ? 0 : (yi >= S ? S - 1 : yi);
        int xi = x + kx - 1; xi = xi < 0 ? 0 : (xi >= S ? S - 1 : xi);
        acc += w[c * 27 + kt * 9 + ky * 3 + kx] * h[(((size_t)(b * T + ti) * S + yi) * S + xi) * 64 + c];
    }
    out[i] = acc;
}
__global__ void dw2_kernel(const float* h, const float* w, float* out, int T, int S, long n) {
    long i = (long)blockIdx.x * blockDim.x + threadIdx.x;
    if (i >= n) return;
    int c = (int)(i & 63); long pos = i >> 6;
    int x = (int)(pos % S); long r = pos / S;
    int y = (int)(r % S); r /= S;
    long bt = r;
    float acc = 0.0f;
    for (int ky = 0; ky < 3; ky++) for (int kx = 0; kx < 3; kx++) {
        int yi = y + ky - 1; yi = yi < 0 ? 0 : (yi >= S ? S - 1 : yi);
        int xi = x + kx - 1; xi = xi < 0 ? 0 : (xi >= S ? S - 1 : xi);
        acc += w[c * 9 + ky * 3 + kx] * h[(((size_t)bt * S + yi) * S + xi) * 64 + c];
    }
    out[i] = acc;
}
__global__ void fmix_kernel(const float* c3, const float* c2, const float* stat,
                            float* F, long n, int Npos) {
    long i = (long)blockIdx.x * blockDim.x + threadIdx.x;
    if (i >= n) return;
    int c = (int)(i & 63); long pos = i >> 6;
    int bc = (int)(pos / Npos) * 64 + c;
    float v1 = fmaxf((c3[i] - stat[bc]) * stat[512 + bc], 0.0f);
    float v2 = fmaxf((c2[i] - stat[1024 + bc]) * stat[1536 + bc], 0.0f);
    F[i] = v1 + v2;
}
__global__ void pool_kernel(const float* F, float* a, int SS, int BT) {
    int i = blockIdx.x * blockDim.x + threadIdx.x;
    if (i >= BT * 64) return;
    int c = i & 63; int bt = i >> 6;
    float s = 0.0f;
    for (int j = 0; j < SS; j++) s += F[((size_t)bt * SS + j) * 64 + c];
    a[i] = s / SS;
}
__global__ void se_kernel(const float* a, const float* aw1, const float* ab1,
                          const float* aw2, const float* ab2, float* a2) {
    __shared__ float sa[64], sh[64];
    int bt = blockIdx.x, o = threadIdx.x;
    sa[o] = a[bt * 64 + o]; __syncthreads();
    float s = ab1[o];
    for (int c = 0; c < 64; c++) s += aw1[o * 64 + c] * sa[c];
    sh[o] = fmaxf(s, 0.0f); __syncthreads();
    s = ab2[o];
    for (int c = 0; c < 64; c++) s += aw2[o * 64 + c] * sh[c];
    a2[bt * 64 + o] = 1.0f / (1.0f + expf(-s));
}
__global__ void fa_kernel(const float* F, const float* a2, unsigned short* out, long n, int SS) {
    long i = (long)blockIdx.x * blockDim.x + threadIdx.x;
    if (i >= n) return;
    int c = (int)(i & 63); long pos = i >> 6;
    int bt = (int)(pos / SS);
    out[i] = f2bf(F[i] * a2[bt * 64 + c]);
}
__global__ void dcsm_out_kernel(const float* r, const float* mR, const float* rR,
                                const float* h, float* out, long n, int Npos) {
    long i = (long)blockIdx.x * blockDim.x + threadIdx.x;
    if (i >= n) return;
    int c = (int)(i & 63); long pos = i >> 6;
    int bc = (int)(pos / Npos) * 64 + c;
    out[i] = fmaxf((r[i] - mR[bc]) * rR[bc], 0.0f) + h[i];
}
__global__ void repack_tc(const float* h, unsigned short* dst, long n) {
    long i = (long)blockIdx.x * blockDim.x + threadIdx.x;
    if (i >= n) return;
    int ch = (int)(i % 640); long r = i / 640;
    int x = (int)(r & 7); r >>= 3;
    int y = (int)(r & 7); int b = (int)(r >> 3);
    int c = ch / 10, t = ch % 10;
    dst[i] = f2bf(h[(((size_t)(b * 10 + t) * 8 + y) * 8 + x) * 64 + c]);
}
__global__ void build_hsse(const float* c3o, float* merged, long n) {
    long i = (long)blockIdx.x * blockDim.x + threadIdx.x;
    if (i >= n) return;
    int dd = (int)(i & 63); long r = i >> 6;
    int lch = (int)(r % 128); int b = (int)(r / 128);
    int y = dd >> 3, x = dd & 7;
    merged[((size_t)b * 256 + lch) * 64 + dd] = c3o[(((size_t)(b * 8 + y) * 8) + x) * 128 + lch];
}
__global__ void build_hlt(const float* z2, float* merged, long n) {
    long i = (long)blockIdx.x * blockDim.x + threadIdx.x;
    if (i >= n) return;
    int dd = (int)(i & 63); long r = i >> 6;
    int ch = (int)(r % 128); int b = (int)(r / 128);
    const float* p = z2 + ((size_t)b * 128 + 2 * dd) * 128 + ch;
    merged[((size_t)b * 256 + 128 + ch) * 64 + dd] = 0.5f * (p[0] + p[128]);
}
__global__ void head_kernel(const float* mixed, const float* w1, const float* b1,
                            const float* w2, const float* b2, const float* w3, const float* b3,
                            float* logits) {
    __shared__ float g[64], h1[32], h2[32];
    int b = blockIdx.x, tid = threadIdx.x;
    float s = 0.0f;
    for (int l = 0; l < 256; l++) s += mixed[((size_t)b * 256 + l) * 64 + tid];
    g[tid] = s / 256.0f; __syncthreads();
    if (tid < 32) {
        float a = b1[tid];
        for (int d2 = 0; d2 < 64; d2++) a += g[d2] * w1[d2 * 32 + tid];
        h1[tid] = act_apply(a, 2);
    }
    __syncthreads();
    if (tid < 32) {
        float a = b2[tid];
        for (int j = 0; j < 32; j++) a += h1[j] * w2[j * 32 + tid];
        h2[tid] = act_apply(a, 2);
    }
    __syncthreads();
    if (tid < 4) {
        float a = b3[tid];
        for (int j = 0; j < 32; j++) a += h2[j] * w3[j * 32 + tid];
        logits[b * 4 + tid] = a;
    }
}

// ---------------- host orchestration ----------------
static inline long cdivl(long a, long b) { return (a + b - 1) / b; }

static void run_gemm(hipStream_t s, const unsigned short* A, const unsigned short* Bm,
                     const float* bias, const float* res, float* outF, unsigned short* outH,
                     long M, int N, int K, int act) {
    long waves = (M / 16) * (N / 64);
    gemm_wmma<<<(int)cdivl(waves, 4), 128, 0, s>>>(A, Bm, bias, res, outF, outH, M, N, K, act);
}

// flat-param indices (jax pytree: dict keys sorted, lists in order)
enum {
    H_B1 = 0, H_B2, H_B3, H_W1, H_W2, H_W3,
    LT_C1B = 6, LT_C1W, LT_C2B, LT_C2W,
    LT_L0 = 10, LT_L1 = 26, MIX = 42,
    SSE_CB1 = 58, SSE_CB2, SSE_CB3, SSE_CW1, SSE_CW2, SSE_CW3,
    DC0 = 64, DC1 = 74, DC2 = 84,
    DSB0 = 94, DSW0 = 97,
    STEM_B = 100, STEM_W = 101,
    ST0 = 102, ST1 = 118, ST2 = 134
};
// ssm sub-offsets: +0 b1 +1 b2 +2 ln1_b +3 ln1_g +4 ln2_b +5 ln2_g
//                  +6 B_im +7 B_re +8 C_im +9 C_re +10 D +11 Lam_im +12 Lam_re +13 log_dt +14 w1 +15 w2
// dcsm sub-offsets: +0 ab1 +1 ab2 +2 aw1 +3 aw2 +4 b2d +5 b3d +6 rb +7 rw +8 w2d +9 w3d

extern "C" void kernel_launch(void* const* d_in, const int* in_sizes, int n_in,
                              void* d_out, int out_size, void* d_ws, size_t ws_size,
                              hipStream_t stream) {
    (void)in_sizes; (void)out_size; (void)ws_size;
    if (n_in < 152) return;
    const float* x_in  = (const float*)d_in[0];
    const float* xhist = (const float*)d_in[1];
    const float* prm[150];
    for (int i = 0; i < 150; i++) prm[i] = (const float*)d_in[2 + i];

    char* base = (char*)d_ws;
    size_t off = 0;
    auto alloc = [&](size_t bytes) -> char* {
        off = (off + 255) & ~(size_t)255;
        char* p = base + off;
        off += bytes;
        return p;
    };

    // packed weights
    unsigned short* Wstem = (unsigned short*)alloc(128 * 64 * 2);
    unsigned short* WdsPk[3];
    for (int l = 0; l < 3; l++) WdsPk[l] = (unsigned short*)alloc(27L * 64 * 64 * 2);
    unsigned short* Wc1Pk = (unsigned short*)alloc(9L * 640 * 256 * 2);
    unsigned short* Wc2Pk = (unsigned short*)alloc(9L * 256 * 128 * 2);
    unsigned short* Wc3Pk = (unsigned short*)alloc(9L * 128 * 128 * 2);
    unsigned short* Wl1Pk = (unsigned short*)alloc(5L * 128 * 128 * 2);
    unsigned short* Wl2Pk = (unsigned short*)alloc(5L * 128 * 128 * 2);
    unsigned short* rwT[3];
    for (int l = 0; l < 3; l++) rwT[l] = (unsigned short*)alloc(64 * 64 * 2);

    struct SsmW { unsigned short *Wbu, *Wc, *W1, *W2; float *lamr, *lami, *fr, *fi; };
    int bases[6] = {ST0, ST1, ST2, LT_L0, LT_L1, MIX};
    int dims[6]  = {64, 64, 64, 128, 128, 64};
    SsmW ssm[6];
    for (int i = 0; i < 6; i++) {
        size_t d = dims[i];
        ssm[i].Wbu  = (unsigned short*)alloc(d * 2 * d * 2);
        ssm[i].Wc   = (unsigned short*)alloc(2 * d * d * 2);
        ssm[i].W1   = (unsigned short*)alloc(d * 2 * d * 2);
        ssm[i].W2   = (unsigned short*)alloc(2 * d * d * 2);
        ssm[i].lamr = (float*)alloc(d * 4);
        ssm[i].lami = (float*)alloc(d * 4);
        ssm[i].fr   = (float*)alloc(d * 4);
        ssm[i].fi   = (float*)alloc(d * 4);
    }

    // activations / temps
    unsigned short* xp    = (unsigned short*)alloc(8L * 10 * 65536 * 4 * 2);
    unsigned short* h0bf  = (unsigned short*)alloc(327680L * 64 * 2);
    float* T1 = (float*)alloc(81920L * 128 * 4);
    float* T2 = (float*)alloc(81920L * 128 * 4);
    float* T3 = (float*)alloc(81920L * 128 * 4);
    float* T4 = (float*)alloc(81920L * 64 * 4);
    float* Xbuf = (float*)alloc(81920L * 64 * 4);
    unsigned short* TB1 = (unsigned short*)alloc(81920L * 128 * 2);
    unsigned short* TB2 = (unsigned short*)alloc(81920L * 128 * 2);
    float* hds   = (float*)alloc(81920L * 64 * 4);
    float* loutF = (float*)alloc(81920L * 64 * 4);
    unsigned short* loutB = (unsigned short*)alloc(81920L * 64 * 2);
    float* stat  = (float*)alloc(4096 * 4);
    float* aP    = (float*)alloc(80L * 64 * 4);
    float* aP2   = (float*)alloc(80L * 64 * 4);
    float* merged = (float*)alloc(8L * 256 * 64 * 4);
    float* ltA = (float*)alloc(4096L * 128 * 4);
    float* ltB = (float*)alloc(4096L * 128 * 4);
    unsigned short* ltBf = (unsigned short*)alloc(4096L * 128 * 2);
    unsigned short* z1bf = (unsigned short*)alloc(2048L * 128 * 2);
    float* z2f = (float*)alloc(1024L * 128 * 4);
    unsigned short* hc2 = (unsigned short*)alloc(512L * 640 * 2);
    unsigned short* c1o = (unsigned short*)alloc(512L * 256 * 2);
    unsigned short* c2o = (unsigned short*)alloc(512L * 128 * 2);
    float* c3o = (float*)alloc(512L * 128 * 4);

    // ---- weight packing / S5 discretization ----
    pack_stem_w<<<(int)cdivl(128 * 64, 256), 256, 0, stream>>>(prm[STEM_W], Wstem);
    pack_x_kernel<<<(int)cdivl(8L * 10 * 65536 * 4, 256), 256, 0, stream>>>(x_in, xp, 8L * 10 * 65536 * 4);
    for (int l = 0; l < 3; l++)
        pack_conv_w<<<(int)cdivl(64L * 64 * 27, 256), 256, 0, stream>>>(prm[DSW0 + l], WdsPk[l], 64, 64, 27);
    pack_conv_w<<<(int)cdivl(256L * 640 * 9, 256), 256, 0, stream>>>(prm[SSE_CW1], Wc1Pk, 256, 640, 9);
    pack_conv_w<<<(int)cdivl(128L * 256 * 9, 256), 256, 0, stream>>>(prm[SSE_CW2], Wc2Pk, 128, 256, 9);
    pack_conv_w<<<(int)cdivl(128L * 128 * 9, 256), 256, 0, stream>>>(prm[SSE_CW3], Wc3Pk, 128, 128, 9);
    pack_conv_w<<<(int)cdivl(128L * 128 * 5, 256), 256, 0, stream>>>(prm[LT_C1W], Wl1Pk, 128, 128, 5);
    pack_conv_w<<<(int)cdivl(128L * 128 * 5, 256), 256, 0, stream>>>(prm[LT_C2W], Wl2Pk, 128, 128, 5);
    for (int l = 0; l < 3; l++)
        transpose_bf<<<(int)cdivl(64 * 64, 256), 256, 0, stream>>>(prm[DC0 + 10 * l + 7], rwT[l], 64, 64);
    for (int i = 0; i < 6; i++) {
        int b = bases[i], d = dims[i];
        s5_lambar<<<1, d, 0, stream>>>(prm[b + 12], prm[b + 11], prm[b + 13],
                                       ssm[i].lamr, ssm[i].lami, ssm[i].fr, ssm[i].fi, d);
        s5_packB<<<(int)cdivl((long)d * d, 256), 256, 0, stream>>>(prm[b + 7], prm[b + 6],
                                                                   ssm[i].fr, ssm[i].fi, ssm[i].Wbu, d);
        s5_packC<<<(int)cdivl((long)d * d, 256), 256, 0, stream>>>(prm[b + 9], prm[b + 8], ssm[i].Wc, d);
        f2bf_copy<<<(int)cdivl((long)d * 2 * d, 256), 256, 0, stream>>>(prm[b + 14], ssm[i].W1, (long)d * 2 * d);
        f2bf_copy<<<(int)cdivl((long)2 * d * d, 256), 256, 0, stream>>>(prm[b + 15], ssm[i].W2, (long)2 * d * d);
    }

    auto run_ssm = [&](int si, long Bb, long Ltok, const float* xin, float* xout, unsigned short* xoutbf) {
        int d = dims[si]; int b = bases[si]; long M = Bb * Ltok;
        SsmW& S = ssm[si];
        ln_kernel<<<(int)M, d, 0, stream>>>(xin, prm[b + 3], prm[b + 2], TB1, d);
        run_gemm(stream, TB1, S.Wbu, nullptr, nullptr, T1, nullptr, M, 2 * d, d, 0);
        s5_scan<<<(int)cdivl(Bb * d, 256), 256, 0, stream>>>(T1, S.lamr, S.lami, TB2, (int)Bb, (int)Ltok, d);
        run_gemm(stream, TB2, S.Wc, nullptr, nullptr, T2, nullptr, M, d, 2 * d, 0);
        s5comb<<<(int)cdivl(M * d, 256), 256, 0, stream>>>(xin, T2, TB1, prm[b + 10], T3, M * d, d);
        ln_kernel<<<(int)M, d, 0, stream>>>(T3, prm[b + 5], prm[b + 4], TB1, d);
        run_gemm(stream, TB1, S.W1, prm[b + 0], nullptr, nullptr, TB2, M, 2 * d, d, 2);
        run_gemm(stream, TB2, S.W2, prm[b + 1], T3, xout, xoutbf, M, d, 2 * d, 0);
    };

    // ---- stem ----
    stem_conv_kernel<<<5120, 128, 0, stream>>>(xp, Wstem, prm[STEM_B], h0bf);

    // ---- SSE stack: ds-conv -> DCSM -> ST-SSM, x3 ----
    const unsigned short* prevBf = h0bf;
    for (int l = 0; l < 3; l++) {
        int Si = 64 >> l, So = 32 >> l;
        long M = 8L * 10 * So * So;
        int Npos = 10 * So * So;
        int SS = So * So;
        int dc = DC0 + 10 * l;
        GeoConv3 g3; g3.T = 10; g3.Si = Si; g3.So = So;
        long waves = M / 16;
        convw_kernel<<<(int)cdivl(waves, 4), 128, 0, stream>>>(g3, prevBf, WdsPk[l], prm[DSB0 + l],
                                                               hds, (unsigned short*)nullptr,
                                                               M, 64, 64, 27, 1, waves);
        dw3_kernel<<<(int)cdivl(M * 64, 256), 256, 0, stream>>>(hds, prm[dc + 9], T1, 10, So, M * 64);
        inorm_stats<<<512, 256, 0, stream>>>(T1, stat + 0, stat + 512, Npos, 64);
        dw2_kernel<<<(int)cdivl(M * 64, 256), 256, 0, stream>>>(hds, prm[dc + 8], T2, 10, So, M * 64);
        inorm_stats<<<512, 256, 0, stream>>>(T2, stat + 1024, stat + 1536, Npos, 64);
        fmix_kernel<<<(int)cdivl(M * 64, 256), 256, 0, stream>>>(T1, T2, stat, T3, M * 64, Npos);
        pool_kernel<<<(int)cdivl(80L * 64, 256), 256, 0, stream>>>(T3, aP, SS, 80);
        se_kernel<<<80, 64, 0, stream>>>(aP, prm[dc + 2], prm[dc + 0], prm[dc + 3], prm[dc + 1], aP2);
        fa_kernel<<<(int)cdivl(M * 64, 256), 256, 0, stream>>>(T3, aP2, TB1, M * 64, SS);
        run_gemm(stream, TB1, rwT[l], prm[dc + 6], nullptr, T4, nullptr, M, 64, 64, 0);
        inorm_stats<<<512, 256, 0, stream>>>(T4, stat + 2048, stat + 2560, Npos, 64);
        dcsm_out_kernel<<<(int)cdivl(M * 64, 256), 256, 0, stream>>>(T4, stat + 2048, stat + 2560,
                                                                     hds, Xbuf, M * 64, Npos);
        run_ssm(l, 8, Npos, Xbuf, loutF, loutB);
        prevBf = loutB;
    }

    // ---- conv2d head -> h_sse rows of merged ----
    repack_tc<<<(int)cdivl(8L * 64 * 640, 256), 256, 0, stream>>>(loutF, hc2, 8L * 64 * 640);
    GeoConv2 g2; g2.S = 8;
    convw_kernel<<<(int)cdivl((512 / 16) * 4, 4), 128, 0, stream>>>(g2, hc2, Wc1Pk, prm[SSE_CB1],
        (float*)nullptr, c1o, 512, 640, 256, 9, 1, (512 / 16) * 4);
    convw_kernel<<<(int)cdivl((512 / 16) * 2, 4), 128, 0, stream>>>(g2, c1o, Wc2Pk, prm[SSE_CB2],
        (float*)nullptr, c2o, 512, 256, 128, 9, 1, (512 / 16) * 2);
    convw_kernel<<<(int)cdivl((512 / 16) * 2, 4), 128, 0, stream>>>(g2, c2o, Wc3Pk, prm[SSE_CB3],
        c3o, (unsigned short*)nullptr, 512, 128, 128, 9, 0, (512 / 16) * 2);
    build_hsse<<<(int)cdivl(8L * 128 * 64, 256), 256, 0, stream>>>(c3o, merged, 8L * 128 * 64);

    // ---- long-range temporal branch ----
    copyf_kernel<<<(int)cdivl(8L * 512 * 128, 256), 256, 0, stream>>>(xhist, ltA, 8L * 512 * 128);
    run_ssm(3, 8, 512, ltA, ltB, nullptr);
    run_ssm(4, 8, 512, ltB, ltA, ltBf);
    GeoConv1 g1a; g1a.Lin = 512; g1a.Lout = 256;
    convw_kernel<<<(int)cdivl((2048 / 16) * 2, 4), 128, 0, stream>>>(g1a, ltBf, Wl1Pk, prm[LT_C1B],
        (float*)nullptr, z1bf, 2048, 128, 128, 5, 1, (2048 / 16) * 2);
    GeoConv1 g1b; g1b.Lin = 256; g1b.Lout = 128;
    convw_kernel<<<(int)cdivl((1024 / 16) * 2, 4), 128, 0, stream>>>(g1b, z1bf, Wl2Pk, prm[LT_C2B],
        z2f, (unsigned short*)nullptr, 1024, 128, 128, 5, 1, (1024 / 16) * 2);
    build_hlt<<<(int)cdivl(8L * 128 * 64, 256), 256, 0, stream>>>(z2f, merged, 8L * 128 * 64);

    // ---- mix SSM (writes `mixed` directly into d_out) + head ----
    float* out_f = (float*)d_out;
    run_ssm(5, 8, 256, merged, out_f + 32, nullptr);
    head_kernel<<<8, 64, 0, stream>>>(out_f + 32, prm[H_W1], prm[H_B1], prm[H_W2], prm[H_B2],
                                      prm[H_W3], prm[H_B3], out_f);
}
